// EnhancedDualHeadTransformerGNN_28630251995630
// MI455X (gfx1250) — matrix operand order repaired and checked by reading, not compiled, not run on hardware
//
#include <hip/hip_runtime.h>

// ---------------------------------------------------------------------------
// EnhancedDualHeadTransformerGNN forward for MI455X (gfx1250, wave32, WMMA).
// - All dense math on v_wmma_f32_16x16x32_bf16 (bf16 A/B, f32 accum).
// - GEMM tiles staged via Tensor Data Mover (tensor_load_to_lds, TENSORcnt),
//   double-buffered in LDS; wave0 drives the DMA, s_wait_tensorcnt + barrier.
// - Attention: fused flash-style WMMA kernel (512 independent 256x256x32).
// - Irregular GAT edge softmax/aggregation via f32 global atomics.
// ---------------------------------------------------------------------------

typedef __attribute__((ext_vector_type(16))) __bf16 v16bf;
typedef __attribute__((ext_vector_type(8)))  float  v8f;
typedef unsigned int u32x4 __attribute__((ext_vector_type(4)));
typedef int          i32x4 __attribute__((ext_vector_type(4)));
typedef int          i32x8 __attribute__((ext_vector_type(8)));

#define NN     16384
#define HH     256
#define GHD    4
#define NODESK 64

#define DEVINL static __device__ __forceinline__

DEVINL unsigned short f2bf(float f) {                 // f32 -> bf16 (RNE)
  unsigned u = __float_as_uint(f);
  unsigned r = u + 0x7FFFu + ((u >> 16) & 1u);
  return (unsigned short)(r >> 16);
}
union BfBits { unsigned short u; __bf16 b; };
DEVINL __bf16 bfu(unsigned short u) { BfBits c; c.u = u; return c.b; }

union FragBF { v16bf v; unsigned u[8]; };             // dword-packed bf16 fragment

// 16-bit A-matrix 16x32 fragment: K offset of element-pair p for lane-half hf
DEVINL int kidx2(int p, int hf) { return ((p >> 2) << 4) + (hf << 3) + ((p & 3) << 1); }

DEVINL float gelu_f(float x) { return 0.5f * x * (1.f + erff(x * 0.70710678118654752f)); }

DEVINL float waveSum(float v) {
#pragma unroll
  for (int m = 16; m > 0; m >>= 1) v += __shfl_xor(v, m, 32);
  return v;
}

// monotonic float<->uint encoding for atomic max on floats
DEVINL unsigned enc_f(float f) { unsigned u = __float_as_uint(f); return (u & 0x80000000u) ? ~u : (u | 0x80000000u); }
DEVINL float    dec_f(unsigned u) { unsigned v = (u & 0x80000000u) ? (u & 0x7FFFFFFFu) : ~u; return __uint_as_float(v); }

DEVINL float blockSum256(float v, float* red) {
  v = waveSum(v);
  const int w = threadIdx.x >> 5, lane = threadIdx.x & 31;
  if (lane == 0) red[w] = v;
  __syncthreads();
  float t = 0.f;
#pragma unroll
  for (int i = 0; i < 8; ++i) t += red[i];
  __syncthreads();
  return t;
}

// ---------------------------------------------------------------------------
// Tensor Data Mover: 2D tile (tile1 rows x tile0 elems, 2-byte data) from
// global (row stride = stride0 elems) into LDS at byte offset lds_off.
// D# layout per CDNA5 ISA 8.3/8.4 (group0: count/lds/global/type=2;
// group1: data_size=2B, tensor dims, tile dims, dim0 stride).
// ---------------------------------------------------------------------------
DEVINL void tdm_load_2d(const void* gptr, unsigned lds_off,
                        unsigned ten0, unsigned ten1,
                        unsigned tile0, unsigned tile1, unsigned stride0) {
  unsigned long long ga = (unsigned long long)gptr;
  u32x4 g0 = { 1u,                                     // count=1, user desc
               lds_off,                                // lds_addr
               (unsigned)(ga & 0xFFFFFFFFu),           // global_addr[31:0]
               (unsigned)((ga >> 32) & 0x1FFFFFFu) | (2u << 30) };  // [56:32] | type=2
  i32x8 g1 = { (int)0x00010000u,                       // data_size=1 (2 bytes)
               (int)((ten0 & 0xFFFFu) << 16),          // tensor_dim0[15:0]
               (int)((ten0 >> 16) | ((ten1 & 0xFFFFu) << 16)),
               (int)((ten1 >> 16) | (tile0 << 16)),    // tensor_dim1 hi | tile_dim0
               (int)(tile1 & 0xFFFFu),                 // tile_dim1, tile_dim2=0
               (int)stride0,                           // tensor_dim0_stride[31:0]
               0, 0 };
  i32x4 gz = { 0, 0, 0, 0 };
#if defined(__clang_major__) && __clang_major__ >= 23
  i32x8 gz8 = { 0, 0, 0, 0, 0, 0, 0, 0 };
  __builtin_amdgcn_tensor_load_to_lds(g0, g1, gz, gz, gz8, 0);
#else
  __builtin_amdgcn_tensor_load_to_lds(g0, g1, gz, gz, 0);
#endif
}

// ---------------------------------------------------------------------------
__global__ __launch_bounds__(256) void k_fill(unsigned* p, unsigned v, int n) {
  int i = blockIdx.x * blockDim.x + threadIdx.x;
  if (i < n) p[i] = v;
}

__global__ __launch_bounds__(256) void k_cast(const float* __restrict__ s, unsigned short* __restrict__ d, int n) {
  int i = blockIdx.x * 256 + threadIdx.x;
  if (i < n) d[i] = f2bf(s[i]);
}

// ---------------------------------------------------------------------------
// C[M,N] = act(A_bf16[M,K] @ Bw_bf16[N,K]^T + bias) + R   (X @ W.T convention)
// FLAGS: 1=bias, 2=gelu, 4=residual-add
// Block tile 128x128x32, 8 waves (2x4), each wave 64x32 (4x2 WMMA frags).
// Tiles staged by TDM into double-buffered dynamic LDS (32 KB):
//   [0,8K)=A par0  [8K,16K)=A par1  [16K,24K)=B par0  [24K,32K)=B par1
// ---------------------------------------------------------------------------
template <int FLAGS>
__global__ __launch_bounds__(256)
void k_gemm(const unsigned short* __restrict__ A, const unsigned short* __restrict__ Bw,
            const float* __restrict__ bias, const float* __restrict__ R,
            float* __restrict__ C, int M, int Nn, int K) {
  extern __shared__ unsigned short smem[];             // dynamic LDS base = offset 0

  const int tid = threadIdx.x;
  const int wave = tid >> 5, lane = tid & 31;
  const int hf = lane >> 4, l16 = lane & 15;
  const int wm = wave >> 2, wn = wave & 3;
  const int bm = blockIdx.x * 128, bn = blockIdx.y * 128;
  const int nk = K >> 5;

  const v8f vz = {0.f, 0.f, 0.f, 0.f, 0.f, 0.f, 0.f, 0.f};
  v8f acc[4][2];
#pragma unroll
  for (int i = 0; i < 4; ++i)
#pragma unroll
    for (int j = 0; j < 2; ++j) acc[i][j] = vz;

  if (wave == 0) {                                     // prime parity-0 tiles
    tdm_load_2d(A + (size_t)bm * K, 0u, (unsigned)K, (unsigned)M, 32u, 128u, (unsigned)K);
    tdm_load_2d(Bw + (size_t)bn * K, 16384u, (unsigned)K, (unsigned)Nn, 32u, 128u, (unsigned)K);
  }

  for (int kb = 0; kb < nk; ++kb) {
    __syncthreads();                                   // prev compute done; safe to DMA next parity
    if (wave == 0) {
      if (kb + 1 < nk) {
        const unsigned np = (unsigned)((kb + 1) & 1);
        tdm_load_2d(A + (size_t)bm * K + ((kb + 1) << 5), np * 8192u,
                    (unsigned)K, (unsigned)M, 32u, 128u, (unsigned)K);
        tdm_load_2d(Bw + (size_t)bn * K + ((kb + 1) << 5), 16384u + np * 8192u,
                    (unsigned)K, (unsigned)Nn, 32u, 128u, (unsigned)K);
        __builtin_amdgcn_s_wait_tensorcnt((short)2);   // current step's 2 tiles resident
      } else {
        __builtin_amdgcn_s_wait_tensorcnt((short)0);
      }
    }
    __syncthreads();                                   // publish LDS tiles

    const unsigned short* As = smem + (kb & 1) * 4096;           // elems, not bytes
    const unsigned short* Bs = smem + 8192 + (kb & 1) * 4096;

    FragBF fa[4];
#pragma unroll
    for (int mt = 0; mt < 4; ++mt) {
      const unsigned short* ap = &As[(wm * 64 + mt * 16 + l16) * 32];
#pragma unroll
      for (int p = 0; p < 8; ++p)
        fa[mt].u[p] = *(const unsigned*)(ap + kidx2(p, hf));
    }
    v16bf bfr[2];
#pragma unroll
    for (int nt = 0; nt < 2; ++nt) {
#pragma unroll
      for (int e = 0; e < 16; ++e) bfr[nt][e] = bfu(Bs[(wn * 32 + nt * 16 + l16) * 32 + ((e >> 3) << 4) + (hf << 3) + (e & 7)]);
    }
#pragma unroll
    for (int mt = 0; mt < 4; ++mt)
#pragma unroll
      for (int nt = 0; nt < 2; ++nt)
        acc[mt][nt] = __builtin_amdgcn_wmma_f32_16x16x32_bf16(
            false, fa[mt].v, false, bfr[nt], (short)0, acc[mt][nt], false, false);
  }

#pragma unroll
  for (int mt = 0; mt < 4; ++mt)
#pragma unroll
    for (int nt = 0; nt < 2; ++nt)
#pragma unroll
      for (int v = 0; v < 8; ++v) {
        int row = bm + wm * 64 + mt * 16 + hf * 8 + v;
        int col = bn + wn * 32 + nt * 16 + l16;
        float x = acc[mt][nt][v];
        if (FLAGS & 1) x += bias[col];
        if (FLAGS & 2) x = gelu_f(x);
        if (FLAGS & 4) x += R[(size_t)row * Nn + col];
        C[(size_t)row * Nn + col] = x;
      }
}

// ---------------------------------------------------------------------------
// GAT pieces
// ---------------------------------------------------------------------------
__global__ __launch_bounds__(256)
void k_gat_scores(const float* __restrict__ hp, const float* __restrict__ as_,
                  const float* __restrict__ ad_, float* __restrict__ alS, float* __restrict__ alD) {
  const int p = blockIdx.x * 8 + (threadIdx.x >> 5);   // (node, head) pair
  const int lane = threadIdx.x & 31;
  const int n = p >> 2, g = p & 3;
  const float* row = hp + (size_t)n * 1024 + g * 256;
  float s1 = 0.f, s2 = 0.f;
  for (int c = lane; c < 256; c += 32) {
    float v = row[c];
    s1 += v * as_[g * 256 + c];
    s2 += v * ad_[g * 256 + c];
  }
  s1 = waveSum(s1); s2 = waveSum(s2);
  if (lane == 0) { alS[p] = s1; alD[p] = s2; }
}

__global__ __launch_bounds__(256)
void k_edge_scores(const int* __restrict__ ei, int E, int E2, const float* __restrict__ alS,
                   const float* __restrict__ alD, float* __restrict__ eb, unsigned* __restrict__ mmax) {
  int ee = blockIdx.x * 256 + threadIdx.x;
  if (ee >= E2) return;
  int s_, d_;
  if (ee < E) { s_ = ei[ee]; d_ = ei[E + ee]; } else { s_ = d_ = ee - E; }
#pragma unroll
  for (int g = 0; g < 4; ++g) {
    float v = alS[s_ * 4 + g] + alD[d_ * 4 + g];
    v = v > 0.f ? v : 0.2f * v;                        // leaky_relu(0.2)
    eb[(size_t)ee * 4 + g] = v;
    atomicMax(&mmax[d_ * 4 + g], enc_f(v));
  }
}

__global__ __launch_bounds__(256)
void k_edge_exp(const int* __restrict__ ei, int E, int E2, float* __restrict__ eb,
                const unsigned* __restrict__ mmax, float* __restrict__ den) {
  int ee = blockIdx.x * 256 + threadIdx.x;
  if (ee >= E2) return;
  int d_;
  if (ee < E) { d_ = ei[E + ee]; } else { d_ = ee - E; }
#pragma unroll
  for (int g = 0; g < 4; ++g) {
    float ex = __expf(eb[(size_t)ee * 4 + g] - dec_f(mmax[d_ * 4 + g]));
    eb[(size_t)ee * 4 + g] = ex;
    atomicAdd(&den[d_ * 4 + g], ex);
  }
}

__global__ __launch_bounds__(256)
void k_edge_aggr(const int* __restrict__ ei, int E, const float* __restrict__ eb,
                 const float* __restrict__ den, const float* __restrict__ hp, float* __restrict__ outb) {
  const int ee = blockIdx.x;
  int s_, d_;
  if (ee < E) { s_ = ei[ee]; d_ = ei[E + ee]; } else { s_ = d_ = ee - E; }
  __shared__ float cf[4];
  if (threadIdx.x < 4)
    cf[threadIdx.x] = 0.25f * eb[(size_t)ee * 4 + threadIdx.x] / den[d_ * 4 + threadIdx.x];
  __syncthreads();
  const int c = threadIdx.x;
  const float* hr = hp + (size_t)s_ * 1024;
  float v = cf[0] * hr[c] + cf[1] * hr[256 + c] + cf[2] * hr[512 + c] + cf[3] * hr[768 + c];
  atomicAdd(&outb[(size_t)d_ * 256 + c], v);
}

__global__ __launch_bounds__(256)
void k_ln_gelu_res(float* __restrict__ h, const float* __restrict__ ob, const float* __restrict__ bias,
                   const float* __restrict__ ga, const float* __restrict__ be) {
  __shared__ float red[8];
  const int row = blockIdx.x, c = threadIdx.x;
  float x = ob[(size_t)row * HH + c] + bias[c];
  float mean = blockSum256(x, red) * (1.f / HH);
  float d = x - mean;
  float var = blockSum256(d * d, red) * (1.f / HH);
  float y = d * rsqrtf(var + 1e-5f) * ga[c] + be[c];
  h[(size_t)row * HH + c] += gelu_f(y);
}

__global__ __launch_bounds__(256)
void k_ln_bf16(const float* __restrict__ t, const float* __restrict__ ga, const float* __restrict__ be,
               unsigned short* __restrict__ dst) {
  __shared__ float red[8];
  const int row = blockIdx.x, c = threadIdx.x;
  float x = t[(size_t)row * HH + c];
  float mean = blockSum256(x, red) * (1.f / HH);
  float d = x - mean;
  float var = blockSum256(d * d, red) * (1.f / HH);
  float y = d * rsqrtf(var + 1e-5f) * ga[c] + be[c];
  dst[(size_t)row * HH + c] = f2bf(y);
}

// ---------------------------------------------------------------------------
// Fused attention: one workgroup per (node-position b, head th).
// seq = 256 graphs, head dim 32. Two-pass flash softmax, all matmuls on WMMA.
// K kept transposed in LDS ([hd][s]); K/V fragment dwords pair two 16-bit loads.
// qkv layout: qkv[(s*64+b)*768 + {0,256,512} + th*32 + hd]
// ---------------------------------------------------------------------------
__global__ __launch_bounds__(256)
void k_attention(const float* __restrict__ qkv, float* __restrict__ o) {
  __shared__ unsigned short Kt[32 * 256];              // [hd][s] transposed
  __shared__ unsigned short Vs[256 * 32];              // [s][hd]
  __shared__ unsigned short Pscr[8][512];              // per-wave 16x32 repack buffer
  const int bq = blockIdx.x & 63;
  const int th = blockIdx.x >> 6;
  const int tid = threadIdx.x, wave = tid >> 5, lane = tid & 31;
  const int hf = lane >> 4, l16 = lane & 15;
  const v8f vz = {0.f, 0.f, 0.f, 0.f, 0.f, 0.f, 0.f, 0.f};

  {
    const int rr = tid;
    const size_t base = (size_t)(rr * NODESK + bq) * 768 + th * 32;
#pragma unroll 4
    for (int hd = 0; hd < 32; ++hd) {
      Kt[hd * 256 + rr] = f2bf(qkv[base + 256 + hd]);
      Vs[rr * 32 + hd] = f2bf(qkv[base + 512 + hd]);
    }
  }
  __syncthreads();

  const float qs = 0.17677669529663687f;               // 1/sqrt(32), folded into Q
  for (int mt = wave; mt < 16; mt += 8) {
    FragBF qa;
    {
      const int m = mt * 16 + l16;
      const size_t base = (size_t)(m * NODESK + bq) * 768 + th * 32;
#pragma unroll
      for (int p = 0; p < 8; ++p) {
        int kk = kidx2(p, hf);
        qa.u[p] = (unsigned)f2bf(qkv[base + kk] * qs) |
                  ((unsigned)f2bf(qkv[base + kk + 1] * qs) << 16);
      }
    }
    // pass 1: row maxes
    float pm[8];
#pragma unroll
    for (int v = 0; v < 8; ++v) pm[v] = -3.0e38f;
    for (int nt = 0; nt < 16; ++nt) {
      FragBF kb;
#pragma unroll
      for (int p = 0; p < 8; ++p) { int kk = kidx2(p, hf); kb.u[p] = (unsigned)Kt[kk * 256 + nt * 16 + l16] | ((unsigned)Kt[(kk + 1) * 256 + nt * 16 + l16] << 16); }
      v8f c = vz;
      c = __builtin_amdgcn_wmma_f32_16x16x32_bf16(false, qa.v, false, kb.v, (short)0, c, false, false);
#pragma unroll
      for (int v = 0; v < 8; ++v) pm[v] = fmaxf(pm[v], c[v]);
    }
#pragma unroll
    for (int m_ = 1; m_ < 16; m_ <<= 1)
#pragma unroll
      for (int v = 0; v < 8; ++v) pm[v] = fmaxf(pm[v], __shfl_xor(pm[v], m_, 32));

    // pass 2: exp, accumulate O = P @ V and row sums
    float rs[8];
#pragma unroll
    for (int v = 0; v < 8; ++v) rs[v] = 0.f;
    v8f oc0 = vz, oc1 = vz;
    for (int np = 0; np < 8; ++np) {
      v8f ca = vz, cb = vz;
      {
        FragBF kb;
#pragma unroll
        for (int p = 0; p < 8; ++p) { int kk = kidx2(p, hf); kb.u[p] = (unsigned)Kt[kk * 256 + (np * 2) * 16 + l16] | ((unsigned)Kt[(kk + 1) * 256 + (np * 2) * 16 + l16] << 16); }
        ca = __builtin_amdgcn_wmma_f32_16x16x32_bf16(false, qa.v, false, kb.v, (short)0, ca, false, false);
      }
      {
        FragBF kb;
#pragma unroll
        for (int p = 0; p < 8; ++p) { int kk = kidx2(p, hf); kb.u[p] = (unsigned)Kt[kk * 256 + (np * 2 + 1) * 16 + l16] | ((unsigned)Kt[(kk + 1) * 256 + (np * 2 + 1) * 16 + l16] << 16); }
        cb = __builtin_amdgcn_wmma_f32_16x16x32_bf16(false, qa.v, false, kb.v, (short)0, cb, false, false);
      }
#pragma unroll
      for (int v = 0; v < 8; ++v) {
        float ea = __expf(ca[v] - pm[v]);
        float eb = __expf(cb[v] - pm[v]);
        rs[v] += ea + eb;
        Pscr[wave][(hf * 8 + v) * 32 + l16] = f2bf(ea);        // C-layout -> A-layout
        Pscr[wave][(hf * 8 + v) * 32 + 16 + l16] = f2bf(eb);
      }
      FragBF pa, vb0, vb1;
#pragma unroll
      for (int p = 0; p < 8; ++p) {
        pa.u[p] = *(const unsigned*)&Pscr[wave][l16 * 32 + kidx2(p, hf)];
        vb0.u[p] = (unsigned)Vs[(np * 32 + kidx2(p, hf)) * 32 + l16] | ((unsigned)Vs[(np * 32 + kidx2(p, hf) + 1) * 32 + l16] << 16);
        vb1.u[p] = (unsigned)Vs[(np * 32 + kidx2(p, hf)) * 32 + 16 + l16] | ((unsigned)Vs[(np * 32 + kidx2(p, hf) + 1) * 32 + 16 + l16] << 16);
      }
      oc0 = __builtin_amdgcn_wmma_f32_16x16x32_bf16(false, pa.v, false, vb0.v, (short)0, oc0, false, false);
      oc1 = __builtin_amdgcn_wmma_f32_16x16x32_bf16(false, pa.v, false, vb1.v, (short)0, oc1, false, false);
    }
#pragma unroll
    for (int m_ = 1; m_ < 16; m_ <<= 1)
#pragma unroll
      for (int v = 0; v < 8; ++v) rs[v] += __shfl_xor(rs[v], m_, 32);
#pragma unroll
    for (int v = 0; v < 8; ++v) {
      float inv = 1.f / rs[v];
      int m = mt * 16 + hf * 8 + v;
      size_t nb = (size_t)(m * NODESK + bq) * HH + th * 32;
      o[nb + l16] = oc0[v] * inv;
      o[nb + 16 + l16] = oc1[v] * inv;
    }
  }
}

// ---------------------------------------------------------------------------
// Pooling + heads
// ---------------------------------------------------------------------------
__global__ __launch_bounds__(256)
void k_pool_score(const float* __restrict__ h, const float* __restrict__ pw,
                  const float* __restrict__ pb, float* __restrict__ sbuf) {
  const int n = blockIdx.x * 8 + (threadIdx.x >> 5), lane = threadIdx.x & 31;
  float s = 0.f;
  for (int c = lane; c < 256; c += 32) s += h[(size_t)n * 256 + c] * pw[c];
  s = waveSum(s);
  if (lane == 0) sbuf[n] = s + pb[0];
}

__global__ __launch_bounds__(256)
void k_gmax(const float* __restrict__ s, unsigned* gm) {
  int i = blockIdx.x * 256 + threadIdx.x;
  float v = s[i];
#pragma unroll
  for (int m = 16; m > 0; m >>= 1) v = fmaxf(v, __shfl_xor(v, m, 32));
  if ((threadIdx.x & 31) == 0) atomicMax(gm, enc_f(v));
}

__global__ __launch_bounds__(256)
void k_gsum(const float* __restrict__ s, const unsigned* __restrict__ gm, float* gs) {
  int i = blockIdx.x * 256 + threadIdx.x;
  float v = __expf(s[i] - dec_f(*gm));
  v = waveSum(v);
  if ((threadIdx.x & 31) == 0) atomicAdd(gs, v);
}

__global__ __launch_bounds__(256)
void k_pool_accum(const float* __restrict__ h, const float* __restrict__ s,
                  const unsigned* __restrict__ gm, const float* __restrict__ gs,
                  float* __restrict__ grep) {
  const int n = blockIdx.x, c = threadIdx.x;
  float aw = __expf(s[n] - dec_f(*gm)) / (*gs);
  atomicAdd(&grep[(n >> 6) * 256 + c], h[(size_t)n * 256 + c] * aw);
}

// ACT: 0 = tanh, 1 = softplus
template <int ACT>
__global__ __launch_bounds__(256)
void k_gemv(const float* __restrict__ A, const float* __restrict__ w,
            const float* __restrict__ b, float* __restrict__ out, int K) {
  const int row = blockIdx.x * 8 + (threadIdx.x >> 5), lane = threadIdx.x & 31;
  float s = 0.f;
  for (int c = lane; c < K; c += 32) s += A[(size_t)row * K + c] * w[c];
  s = waveSum(s);
  if (lane == 0) {
    s += b[0];
    out[row] = (ACT == 0) ? tanhf(s) : (s > 20.f ? s : log1pf(__expf(s)));
  }
}

// ---------------------------------------------------------------------------
extern "C" void kernel_launch(void* const* d_in, const int* in_sizes, int n_in,
                              void* d_out, int out_size, void* d_ws, size_t ws_size,
                              hipStream_t stream) {
  (void)n_in; (void)out_size; (void)ws_size;
  const float* x      = (const float*)d_in[0];
  const int*   ei     = (const int*)  d_in[1];
  const float* emb_w  = (const float*)d_in[4];
  const float* emb_b  = (const float*)d_in[5];
  const float* gat_w  = (const float*)d_in[6];
  const float* gat_as = (const float*)d_in[7];
  const float* gat_ad = (const float*)d_in[8];
  const float* gat_b  = (const float*)d_in[9];
  const float* gnn_g  = (const float*)d_in[10];
  const float* gnn_be = (const float*)d_in[11];
  const float* ln1_g  = (const float*)d_in[12];
  const float* ln1_b  = (const float*)d_in[13];
  const float* in_w   = (const float*)d_in[14];
  const float* in_b   = (const float*)d_in[15];
  const float* out_w  = (const float*)d_in[16];
  const float* out_b  = (const float*)d_in[17];
  const float* ln2_g  = (const float*)d_in[18];
  const float* ln2_b  = (const float*)d_in[19];
  const float* ff1_w  = (const float*)d_in[20];
  const float* ff1_b  = (const float*)d_in[21];
  const float* ff2_w  = (const float*)d_in[22];
  const float* ff2_b  = (const float*)d_in[23];
  const float* pool_w = (const float*)d_in[24];
  const float* pool_b = (const float*)d_in[25];
  const float* vw1    = (const float*)d_in[26];
  const float* vb1    = (const float*)d_in[27];
  const float* vw2    = (const float*)d_in[28];
  const float* vb2    = (const float*)d_in[29];
  const float* pw1    = (const float*)d_in[30];
  const float* pb1    = (const float*)d_in[31];
  const float* pw2    = (const float*)d_in[32];
  const float* pb2    = (const float*)d_in[33];
  const float* uw1    = (const float*)d_in[34];
  const float* ub1    = (const float*)d_in[35];
  const float* uw2    = (const float*)d_in[36];
  const float* ub2    = (const float*)d_in[37];

  const int E  = in_sizes[1] / 2;
  const int E2 = E + NN;
  const unsigned GSM = 32768;                          // k_gemm dynamic LDS

  char* W = (char*)d_ws;
  size_t off = 0;
  auto alloc = [&](size_t bytes) -> char* {
    char* p = W + off;
    off += (bytes + 255) & ~(size_t)255;
    return p;
  };
  float*          hpA  = (float*)alloc((size_t)NN * 1024 * 4);            // hp / qkv / ff1
  float*          hB   = (float*)alloc((size_t)NN * 256 * 4);             // h / t (persistent)
  float*          outC = (float*)alloc((size_t)NN * 256 * 4);             // GAT out / attn o / tmp
  unsigned short* bfA  = (unsigned short*)alloc((size_t)NN * 1024 * 2);   // bf16 activations
  unsigned short* bfW  = (unsigned short*)alloc((size_t)1024 * 1024 * 2); // bf16 weights
  float*          ebuf = (float*)alloc((size_t)E2 * 4 * 4);
  unsigned*       mmax = (unsigned*)alloc((size_t)NN * 4 * 4);
  float*          den  = (float*)alloc((size_t)NN * 4 * 4);
  float*          alS  = (float*)alloc((size_t)NN * 4 * 4);
  float*          alD  = (float*)alloc((size_t)NN * 4 * 4);
  float*          sbuf = (float*)alloc((size_t)NN * 4);
  unsigned*       gmaxp= (unsigned*)alloc(256);
  float*          gsump= (float*)alloc(256);
  float*          grep = (float*)alloc(256 * 256 * 4);
  float*          vtmp = (float*)alloc(256 * 256 * 4);
  float*          utmp = (float*)alloc(256 * 128 * 4);

  float* d_val = (float*)d_out;
  float* d_pol = d_val + 256;
  float* d_unc = d_pol + (size_t)NN * 256;

  // ---- embedding: h = x @ emb_w.T + emb_b ----
  k_cast<<<(NN * 128) / 256, 256, 0, stream>>>(x, bfA, NN * 128);
  k_cast<<<(256 * 128) / 256, 256, 0, stream>>>(emb_w, bfW, 256 * 128);
  k_gemm<1><<<dim3(NN / 128, 2), 256, GSM, stream>>>(bfA, bfW, emb_b, nullptr, hB, NN, 256, 128);

  // ---- 8 residual GAT layers ----
  for (int l = 0; l < 8; ++l) {
    k_cast<<<(NN * 256) / 256, 256, 0, stream>>>(hB, bfA, NN * 256);
    k_cast<<<(1024 * 256) / 256, 256, 0, stream>>>(gat_w + (size_t)l * 1024 * 256, bfW, 1024 * 256);
    k_gemm<0><<<dim3(NN / 128, 8), 256, GSM, stream>>>(bfA, bfW, nullptr, nullptr, hpA, NN, 1024, 256);
    k_gat_scores<<<NN * GHD / 8, 256, 0, stream>>>(hpA, gat_as + l * GHD * 256, gat_ad + l * GHD * 256, alS, alD);
    k_fill<<<(NN * 4) / 256, 256, 0, stream>>>(mmax, 0u, NN * 4);
    k_fill<<<(NN * 4) / 256, 256, 0, stream>>>((unsigned*)den, 0u, NN * 4);
    k_fill<<<(NN * 256) / 256, 256, 0, stream>>>((unsigned*)outC, 0u, NN * 256);
    k_edge_scores<<<(E2 + 255) / 256, 256, 0, stream>>>(ei, E, E2, alS, alD, ebuf, mmax);
    k_edge_exp<<<(E2 + 255) / 256, 256, 0, stream>>>(ei, E, E2, ebuf, mmax, den);
    k_edge_aggr<<<E2, 256, 0, stream>>>(ei, E, ebuf, den, hpA, outC);
    k_ln_gelu_res<<<NN, 256, 0, stream>>>(hB, outC, gat_b + l * 256, gnn_g + l * 256, gnn_be + l * 256);
  }

  // ---- 6 transformer layers ----
  for (int l = 0; l < 6; ++l) {
    k_ln_bf16<<<NN, 256, 0, stream>>>(hB, ln1_g + l * 256, ln1_b + l * 256, bfA);
    k_cast<<<(768 * 256) / 256, 256, 0, stream>>>(in_w + (size_t)l * 768 * 256, bfW, 768 * 256);
    k_gemm<1><<<dim3(NN / 128, 6), 256, GSM, stream>>>(bfA, bfW, in_b + l * 768, nullptr, hpA, NN, 768, 256);
    k_attention<<<512, 256, 0, stream>>>(hpA, outC);
    k_cast<<<(NN * 256) / 256, 256, 0, stream>>>(outC, bfA, NN * 256);
    k_cast<<<(256 * 256) / 256, 256, 0, stream>>>(out_w + (size_t)l * 256 * 256, bfW, 256 * 256);
    k_gemm<5><<<dim3(NN / 128, 2), 256, GSM, stream>>>(bfA, bfW, out_b + l * 256, hB, hB, NN, 256, 256);
    k_ln_bf16<<<NN, 256, 0, stream>>>(hB, ln2_g + l * 256, ln2_b + l * 256, bfA);
    k_cast<<<(1024 * 256) / 256, 256, 0, stream>>>(ff1_w + (size_t)l * 1024 * 256, bfW, 1024 * 256);
    k_gemm<3><<<dim3(NN / 128, 8), 256, GSM, stream>>>(bfA, bfW, ff1_b + l * 1024, nullptr, hpA, NN, 1024, 256);
    k_cast<<<(NN * 1024) / 256, 256, 0, stream>>>(hpA, bfA, NN * 1024);
    k_cast<<<(256 * 1024) / 256, 256, 0, stream>>>(ff2_w + (size_t)l * 256 * 1024, bfW, 256 * 1024);
    k_gemm<5><<<dim3(NN / 128, 2), 256, GSM, stream>>>(bfA, bfW, ff2_b + l * 256, hB, hB, NN, 256, 1024);
  }

  // ---- global attention pooling ----
  k_pool_score<<<NN / 8, 256, 0, stream>>>(hB, pool_w, pool_b, sbuf);
  k_fill<<<1, 256, 0, stream>>>(gmaxp, 0u, 1);
  k_fill<<<1, 256, 0, stream>>>((unsigned*)gsump, 0u, 1);
  k_fill<<<(256 * 256) / 256, 256, 0, stream>>>((unsigned*)grep, 0u, 256 * 256);
  k_gmax<<<NN / 256, 256, 0, stream>>>(sbuf, gmaxp);
  k_gsum<<<NN / 256, 256, 0, stream>>>(sbuf, gmaxp, gsump);
  k_pool_accum<<<NN, 256, 0, stream>>>(hB, sbuf, gmaxp, gsump, grep);

  // ---- value head ----
  k_cast<<<(256 * 256) / 256, 256, 0, stream>>>(grep, bfA, 256 * 256);
  k_cast<<<(256 * 256) / 256, 256, 0, stream>>>(vw1, bfW, 256 * 256);
  k_gemm<3><<<dim3(2, 2), 256, GSM, stream>>>(bfA, bfW, vb1, nullptr, vtmp, 256, 256, 256);
  k_gemv<0><<<256 / 8, 256, 0, stream>>>(vtmp, vw2, vb2, d_val, 256);

  // ---- uncertainty head ----
  k_cast<<<(128 * 256) / 256, 256, 0, stream>>>(uw1, bfW, 128 * 256);
  k_gemm<3><<<dim3(2, 1), 256, GSM, stream>>>(bfA, bfW, ub1, nullptr, utmp, 256, 128, 256);
  k_gemv<1><<<256 / 8, 256, 0, stream>>>(utmp, uw2, ub2, d_unc, 128);

  // ---- policy head ----
  k_cast<<<(NN * 256) / 256, 256, 0, stream>>>(hB, bfA, NN * 256);
  k_cast<<<(256 * 256) / 256, 256, 0, stream>>>(pw1, bfW, 256 * 256);
  k_gemm<3><<<dim3(NN / 128, 2), 256, GSM, stream>>>(bfA, bfW, pb1, nullptr, outC, NN, 256, 256);
  k_cast<<<(NN * 256) / 256, 256, 0, stream>>>(outC, bfA, NN * 256);
  k_cast<<<(256 * 256) / 256, 256, 0, stream>>>(pw2, bfW, 256 * 256);
  k_gemm<1><<<dim3(NN / 128, 2), 256, GSM, stream>>>(bfA, bfW, pb2, nullptr, d_pol, NN, 256, 256);
}
